// SwinTransformerBlock_49203145343525
// MI455X (gfx1250) — compile-verified, matrix operand order checked by reading, not run on hardware
//
#include <hip/hip_runtime.h>
#include <hip/hip_bf16.h>
#include <math.h>
#include <stdint.h>

// ---------------- problem constants ----------------
#define B_IMG   4
#define H_IMG   128
#define W_IMG   128
#define CC      256
#define NHEAD   8
#define WSZ     8
#define SSH     4
#define HD      32
#define NWIN    1024          // B * (128/8) * (128/8)
#define NTOK    (H_IMG*W_IMG) // 16384
#define MTOT    (B_IMG*NTOK)  // 65536 tokens total
#define SROWS   (NWIN*128)    // 131072 rows of s = [x_w | y_w]

typedef __attribute__((ext_vector_type(16))) _Float16 v16h;
typedef __attribute__((ext_vector_type(8)))  float    v8f;
typedef __attribute__((ext_vector_type(2)))  _Float16 h2v;

// ---------------- WMMA fragment helpers (CDNA5 16-bit layouts) ----------------
// A (16x32, MxK): lane L holds row M = L%16 ; VGPR v holds K pair:
//   base = (L<16 ? 0 : 8) + (v>=4 ? 16 : 0), k = base + (v&3)*2 , k+1
__device__ __forceinline__ int frag_k(int v, int lane) {
  return (lane < 16 ? 0 : 8) + ((v & 4) ? 16 : 0) + ((v & 3) << 1);
}

// Load a 16x32 f16 fragment where element (r, k) lives at p[(r0+r)*ld + k0 + k]
// (used for A row-major, and for B when B[k][n] = mem[n*ld + k], i.e. weights W[n][k]).
__device__ __forceinline__ v16h load_frag_rows(const _Float16* __restrict__ p,
                                               int r0, int ld, int k0, int lane) {
  v16h f;
  const _Float16* rp = p + (long long)(r0 + (lane & 15)) * ld + k0;
#pragma unroll
  for (int v = 0; v < 8; ++v) {
    int k = frag_k(v, lane);
    h2v pr = *(const h2v*)(rp + k);   // paired K values are contiguous -> 32-bit load
    f[2 * v] = pr[0];
    f[2 * v + 1] = pr[1];
  }
  return f;
}

__device__ __forceinline__ v8f wmma16(v16h a, v16h b, v8f c) {
  // v_wmma_f32_16x16x32_f16  D = A*B + C
  return __builtin_amdgcn_wmma_f32_16x16x32_f16(false, a, false, b, (short)0, c,
                                                false, false);
}

__device__ __forceinline__ float wave_sum(float v) {
#pragma unroll
  for (int o = 16; o > 0; o >>= 1) v += __shfl_xor(v, o, 32);
  return v;
}

// ---------------- CDNA5 async global->LDS copy (ASYNCcnt path) ----------------
// GV mode: per-lane 16B from global address (VGPR pair) into LDS offset (VGPR).
__device__ __forceinline__ void async_ld_b128(unsigned lds_off, const void* g) {
  asm volatile("global_load_async_to_lds_b128 %0, %1, off"
               :: "v"(lds_off), "v"((unsigned long long)(uintptr_t)g)
               : "memory");
}
template <int N>
__device__ __forceinline__ void wait_asynccnt() {
  asm volatile("s_wait_asynccnt %0" :: "i"(N) : "memory");
}

// ---------------- weight f32 -> f16 ----------------
__global__ void cvt_f16_kernel(const float* __restrict__ src,
                               _Float16* __restrict__ dst, int n) {
  int i = blockIdx.x * 256 + threadIdx.x;
  if (i < n) dst[i] = (_Float16)src[i];
}

// ---------------- LN1 + roll + window partition ----------------
// One wave32 per (window, slot) token: slot 0..63 -> x_w, 64..127 -> y_w.
// Writes s_buf[win*128 + slot][c] (f16) and per-query mask scalars.
__global__ __launch_bounds__(256)
void ln_part_kernel(const float* __restrict__ x, const float* __restrict__ y,
                    const float* __restrict__ qm, const float* __restrict__ sm,
                    const float* __restrict__ pp, const float* __restrict__ g,
                    const float* __restrict__ bb,
                    _Float16* __restrict__ s_buf, float* __restrict__ qb_s,
                    float* __restrict__ sb_s, float* __restrict__ ppc_s,
                    float* __restrict__ hor_s) {
  int slot = blockIdx.x * 8 + (threadIdx.x >> 5);   // 0 .. SROWS-1
  int lane = threadIdx.x & 31;
  int win = slot >> 7, t = slot & 127;
  int bimg = win >> 8, wrem = win & 255;
  int wh = wrem >> 4, ww = wrem & 15;
  int tt = t & 63;
  int ri = wh * WSZ + (tt >> 3), rj = ww * WSZ + (tt & 7);
  int oi = (ri + SSH) & (H_IMG - 1), oj = (rj + SSH) & (W_IMG - 1);
  long long tok = (long long)bimg * NTOK + oi * W_IMG + oj;
  const float* base = ((t < 64) ? x : y) + tok * CC;

  float v[8], s = 0.f;
#pragma unroll
  for (int i = 0; i < 8; ++i) { v[i] = base[i * 32 + lane]; s += v[i]; }
  float mu = wave_sum(s) * (1.0f / CC);
  float vs = 0.f;
#pragma unroll
  for (int i = 0; i < 8; ++i) { float d = v[i] - mu; vs += d * d; }
  float inv = rsqrtf(wave_sum(vs) * (1.0f / CC) + 1e-5f);
  _Float16* dst = s_buf + (long long)slot * CC;
#pragma unroll
  for (int i = 0; i < 8; ++i) {
    int c = i * 32 + lane;
    dst[c] = (_Float16)((v[i] - mu) * inv * g[c] + bb[c]);
  }
  if (t < 64 && lane == 0) {
    long long q = (long long)win * 64 + t;
    qb_s[q] = (qm[tok] >= 0.5f) ? 1.0f : 0.0f;
    float sv = (sm[tok] >= 0.5f) ? 1.0f : 0.0f;
    sb_s[q] = sv;
    float pc = pp[tok]; pc = (pc >= 1.0f) ? 1.0f : pc;
    ppc_s[q] = pc;
    hor_s[q] = (pc < 0.3f) ? 0.0f : ((pc > 0.7f) ? 1.0f : pc);
  }
}

// ---------------- generic WMMA GEMM: out = act(A * W^T + bias [+resid]) ----------------
// A: f16 [M,K] row-major. W: f16 [Nn,K] row-major (so B[k][n] = W[n][k]).
// Block = 256 thr = 8 waves; block tile 64x128; wave tile 16x64 (4 accumulators).
// Weight tiles (shared by 4 waves each) are double-buffer staged into LDS with
// CDNA5 async global->LDS copies; WMMA of step k overlaps the copy of step k+1.
template <bool OUTF16, bool GELU, bool RESID>
__global__ __launch_bounds__(256)
void gemm_wmma(const _Float16* __restrict__ A, const _Float16* __restrict__ Wt,
               const float* __restrict__ bias, const float* __restrict__ resid,
               void* __restrict__ outp, int M, int Nn, int K) {
  __shared__ _Float16 Bs[2][128][32];   // ping-pong 128(N) x 32(K) weight tiles

  const int tid = threadIdx.x;
  const int lane = tid & 31;
  const int wave = tid >> 5;
  const int m0 = blockIdx.x * 64 + (wave >> 1) * 16;
  const int bn = blockIdx.y * 128 + (wave & 1) * 64;
  const int gbn = blockIdx.y * 128;

  // stage one 128x32 weight tile: 512 x 16B chunks, 2 per thread
  auto stage = [&](int buf, int k0) {
#pragma unroll
    for (int c = 0; c < 2; ++c) {
      int cid = tid + c * 256;          // 0..511
      int n = cid >> 2;                 // 0..127
      int ko = (cid & 3) * 8;           // half offset within the 32-wide K slab
      const void* src = Wt + (long long)(gbn + n) * K + (k0 + ko);
      unsigned dst = (unsigned)(uintptr_t)&Bs[buf][n][ko];
      async_ld_b128(dst, src);
    }
  };

  const int nk = K / 32;
  v8f acc[4] = {};
  stage(0, 0);
  for (int kk = 0; kk < nk; ++kk) {
    const int cur = kk & 1;
    if (kk + 1 < nk) { stage(cur ^ 1, (kk + 1) * 32); wait_asynccnt<2>(); }
    else             { wait_asynccnt<0>(); }
    __syncthreads();                    // tile `cur` staged by all waves

    v16h a = load_frag_rows(A, m0, K, kk * 32, lane);
#pragma unroll
    for (int j = 0; j < 4; ++j) {
      int nloc = (wave & 1) * 64 + j * 16 + (lane & 15);
      v16h b;
#pragma unroll
      for (int v = 0; v < 8; ++v) {     // B fragment from LDS (ds_load path)
        int k = frag_k(v, lane);
        b[2 * v] = Bs[cur][nloc][k];
        b[2 * v + 1] = Bs[cur][nloc][k + 1];
      }
      acc[j] = wmma16(a, b, acc[j]);
    }
    __syncthreads();                    // done reading `cur` before its reuse
  }

  const int mb = m0 + ((lane < 16) ? 0 : 8);
#pragma unroll
  for (int j = 0; j < 4; ++j) {
    int n = bn + j * 16 + (lane & 15);
    float bv = bias[n];
#pragma unroll
    for (int v = 0; v < 8; ++v) {
      float val = acc[j][v] + bv;
      if (GELU) val = 0.5f * val * (1.0f + erff(val * 0.70710678118654752f));
      long long idx = (long long)(mb + v) * Nn + n;
      if (RESID) val += resid[idx];
      if (OUTF16) ((_Float16*)outp)[idx] = (_Float16)val;
      else        ((float*)outp)[idx] = val;
    }
  }
}

// ---------------- per-(window, head) attention ----------------
// scores (64x128) via WMMA, masked self/cross epilogue, softmax, P@V via WMMA.
__global__ __launch_bounds__(256)
void attn_kernel(const _Float16* __restrict__ qbuf,
                 const _Float16* __restrict__ kvbuf,
                 const float* __restrict__ qb_s, const float* __restrict__ sb_s,
                 const float* __restrict__ ppc_s, const float* __restrict__ hor_s,
                 const float* __restrict__ wpq, const float* __restrict__ bpq,
                 const float* __restrict__ wps, const float* __restrict__ bps,
                 _Float16* __restrict__ attn_out) {
  const int win = blockIdx.x, head = blockIdx.y;
  const int tid = threadIdx.x, lane = tid & 31, wave = tid >> 5;

  __shared__ float   sc[64][130];          // scores (padded)
  __shared__ _Float16 pbuf[64][128];       // softmax probs, f16 for WMMA
  __shared__ float qn[64], kn[64], qbv[64], sbv[64], ppv[64], hov[64], S4[4];

  const _Float16* Qb = qbuf + (long long)win * 128 * CC + head * HD;
  const _Float16* Kb = kvbuf + (long long)win * 128 * (2 * CC) + head * HD;
  const _Float16* Vb = kvbuf + (long long)win * 128 * (2 * CC) + CC + head * HD;

  // per-head rank-1 projector scalars: qp.sp = pp*sb*Sws + pp*Swb + sb*Sbw + Sbb
  if (wave == 0) {
    int c = head * HD + lane;                 // lane = d, 0..31
    float a = wpq[c] * wps[c], b = wpq[c] * bps[c];
    float c2 = bpq[c] * wps[c], d2 = bpq[c] * bps[c];
    a = wave_sum(a); b = wave_sum(b); c2 = wave_sum(c2); d2 = wave_sum(d2);
    if (lane == 0) { S4[0] = a; S4[1] = b; S4[2] = c2; S4[3] = d2; }
  }
  if (tid < 64) {                             // query norms + per-query scalars
    float s = 0.f;
    for (int d = 0; d < HD; ++d) { float q = (float)Qb[tid * CC + d]; s += q * q; }
    qn[tid] = sqrtf(s);
    long long qi = (long long)win * 64 + tid;
    qbv[tid] = qb_s[qi]; sbv[tid] = sb_s[qi]; ppv[tid] = ppc_s[qi]; hov[tid] = hor_s[qi];
  } else if (tid < 128) {                     // cross-key norms (y half)
    int m = tid - 64;
    float s = 0.f;
    for (int d = 0; d < HD; ++d) { float k = (float)Kb[(64 + m) * (2 * CC) + d]; s += k * k; }
    kn[m] = sqrtf(s);
  }
  __syncthreads();

  // ---- scores = Q @ K^T (M=64, N=128, K=32 -> single WMMA step per tile) ----
  {
    int mt = wave >> 1, ntb = (wave & 1) * 4;
    v16h a = load_frag_rows(Qb, mt * 16, CC, 0, lane);
#pragma unroll
    for (int j = 0; j < 4; ++j) {
      int n0 = (ntb + j) * 16;
      v16h b = load_frag_rows(Kb, n0, 2 * CC, 0, lane);  // B[k=d][n=key]=K[key][d]
      v8f d = {};
      d = wmma16(a, b, d);
      int n = n0 + (lane & 15);
      int mb = mt * 16 + ((lane < 16) ? 0 : 8);
#pragma unroll
      for (int v = 0; v < 8; ++v) sc[mb + v][n] = d[v];
    }
  }
  __syncthreads();

  // ---- masking / cross normalization ----
  for (int idx = tid; idx < 64 * 128; idx += 256) {
    int r = idx >> 7, m = idx & 127;
    float raw = sc[r][m], val;
    if (m < 64) {                                        // self part
      val = raw * 0.17677669529663687f;                  // HD^-0.5
      if (qbv[r] + qbv[m] == 1.0f) val = -__builtin_inff();
    } else {                                             // cross part
      int mc = m - 64;
      float dot = ppv[r] * sbv[mc] * S4[0] + ppv[r] * S4[1] + sbv[mc] * S4[2] + S4[3];
      val = dot * (raw / (qn[r] * kn[mc] + 1e-7f));
      if (hov[r] + sbv[mc] == 1.0f) val = -__builtin_inff();
    }
    sc[r][m] = val;
  }
  __syncthreads();

  // ---- softmax rows ----
  if (tid < 64) {
    float mx = -__builtin_inff();
    for (int m = 0; m < 128; ++m) mx = fmaxf(mx, sc[tid][m]);
    float sum = 0.f;
    for (int m = 0; m < 128; ++m) { float e = __expf(sc[tid][m] - mx); sum += e; sc[tid][m] = e; }
    float inv = 1.0f / sum;
    for (int m = 0; m < 128; ++m) pbuf[tid][m] = (_Float16)(sc[tid][m] * inv);
  }
  __syncthreads();

  // ---- out = P @ V (M=64, N=32, K=128 -> 4 WMMA steps), 8 tiles / 8 waves ----
  {
    int mt = wave & 3, nt = wave >> 2;
    v8f acc = {};
    for (int kk = 0; kk < 4; ++kk) {
      int k0 = kk * 32;
      v16h a, b;
      int r = mt * 16 + (lane & 15);
      int n = nt * 16 + (lane & 15);
#pragma unroll
      for (int v = 0; v < 8; ++v) {            // A from LDS probs (ds_load path)
        int k = k0 + frag_k(v, lane);
        a[2 * v] = pbuf[r][k];
        a[2 * v + 1] = pbuf[r][k + 1];
      }
#pragma unroll
      for (int v = 0; v < 8; ++v) {            // B[k=key][n=d] = V[key][d] (col-major pairs)
        int k = k0 + frag_k(v, lane);
        b[2 * v] = Vb[(long long)k * (2 * CC) + n];
        b[2 * v + 1] = Vb[(long long)(k + 1) * (2 * CC) + n];
      }
      acc = wmma16(a, b, acc);
    }
    int n = nt * 16 + (lane & 15);
    int mb = mt * 16 + ((lane < 16) ? 0 : 8);
#pragma unroll
    for (int v = 0; v < 8; ++v)
      attn_out[((long long)win * 64 + mb + v) * CC + head * HD + n] = (_Float16)acc[v];
  }
}

// ---------------- window reverse + unroll + residual + LN2 ----------------
__global__ __launch_bounds__(256)
void fuse_resid_ln2(const float* __restrict__ x, const float* __restrict__ proj,
                    const float* __restrict__ g2, const float* __restrict__ b2,
                    float* __restrict__ x1, _Float16* __restrict__ xm) {
  int slot = blockIdx.x * 8 + (threadIdx.x >> 5);   // token 0..MTOT-1
  int lane = threadIdx.x & 31;
  int bimg = slot >> 14, hw = slot & (NTOK - 1);
  int i = hw >> 7, j = hw & (W_IMG - 1);
  int ri = (i - SSH) & (H_IMG - 1), rj = (j - SSH) & (W_IMG - 1);
  int win = bimg * 256 + (ri >> 3) * 16 + (rj >> 3);
  int t = (ri & 7) * 8 + (rj & 7);
  const float* xr = x + (long long)slot * CC;
  const float* pr = proj + ((long long)win * 64 + t) * CC;
  float* x1r = x1 + (long long)slot * CC;

  float v[8], s = 0.f;
#pragma unroll
  for (int q = 0; q < 8; ++q) {
    int c = q * 32 + lane;
    v[q] = xr[c] + pr[c];
    x1r[c] = v[q];
    s += v[q];
  }
  float mu = wave_sum(s) * (1.0f / CC);
  float vs = 0.f;
#pragma unroll
  for (int q = 0; q < 8; ++q) { float d = v[q] - mu; vs += d * d; }
  float inv = rsqrtf(wave_sum(vs) * (1.0f / CC) + 1e-5f);
  _Float16* xmr = xm + (long long)slot * CC;
#pragma unroll
  for (int q = 0; q < 8; ++q) {
    int c = q * 32 + lane;
    xmr[c] = (_Float16)((v[q] - mu) * inv * g2[c] + b2[c]);
  }
}

// ---------------- host-side orchestration ----------------
extern "C" void kernel_launch(void* const* d_in, const int* in_sizes, int n_in,
                              void* d_out, int out_size, void* d_ws, size_t ws_size,
                              hipStream_t stream) {
  const float* x       = (const float*)d_in[0];
  const float* y       = (const float*)d_in[1];
  const float* q_mask  = (const float*)d_in[2];
  const float* s_mask  = (const float*)d_in[3];
  const float* pro_pred= (const float*)d_in[4];
  const float* norm1_g = (const float*)d_in[5];
  const float* norm1_b = (const float*)d_in[6];
  const float* wq      = (const float*)d_in[7];
  const float* bq      = (const float*)d_in[8];
  const float* wkv     = (const float*)d_in[9];
  const float* bkv     = (const float*)d_in[10];
  const float* w_proj  = (const float*)d_in[11];
  const float* b_proj  = (const float*)d_in[12];
  const float* w_pq    = (const float*)d_in[13];
  const float* b_pq    = (const float*)d_in[14];
  const float* w_ps    = (const float*)d_in[15];
  const float* b_ps    = (const float*)d_in[16];
  const float* norm2_g = (const float*)d_in[17];
  const float* norm2_b = (const float*)d_in[18];
  const float* w_fc1   = (const float*)d_in[19];
  const float* b_fc1   = (const float*)d_in[20];
  const float* w_fc2   = (const float*)d_in[21];
  const float* b_fc2   = (const float*)d_in[22];

  // ---- workspace carve ----
  size_t off = 0;
  auto carve = [&](size_t bytes) -> void* {
    void* p = (char*)d_ws + off;
    off += (bytes + 255) & ~(size_t)255;
    return p;
  };
  _Float16* s_buf  = (_Float16*)carve((size_t)SROWS * CC * 2);          // 64 MB
  _Float16* q_buf  = (_Float16*)carve((size_t)SROWS * CC * 2);          // 64 MB
  _Float16* kv_buf = (_Float16*)carve((size_t)SROWS * 2 * CC * 2);      // 128 MB
  _Float16* attn_o = (_Float16*)carve((size_t)MTOT * CC * 2);           // 32 MB
  float*    proj_o = (float*)   carve((size_t)MTOT * CC * 4);           // 64 MB
  float*    x1     = (float*)   carve((size_t)MTOT * CC * 4);           // 64 MB
  _Float16* xm     = (_Float16*)carve((size_t)MTOT * CC * 2);           // 32 MB
  _Float16* h_buf  = (_Float16*)carve((size_t)MTOT * 4 * CC * 2);       // 128 MB
  float*    qb_s   = (float*)carve((size_t)MTOT * 4);
  float*    sb_s   = (float*)carve((size_t)MTOT * 4);
  float*    ppc_s  = (float*)carve((size_t)MTOT * 4);
  float*    hor_s  = (float*)carve((size_t)MTOT * 4);
  _Float16* wq_h   = (_Float16*)carve((size_t)CC * CC * 2);
  _Float16* wkv_h  = (_Float16*)carve((size_t)2 * CC * CC * 2);
  _Float16* wpr_h  = (_Float16*)carve((size_t)CC * CC * 2);
  _Float16* wf1_h  = (_Float16*)carve((size_t)4 * CC * CC * 2);
  _Float16* wf2_h  = (_Float16*)carve((size_t)4 * CC * CC * 2);

  // ---- weight conversion ----
  auto cvt = [&](const float* s, _Float16* d, int n) {
    cvt_f16_kernel<<<(n + 255) / 256, 256, 0, stream>>>(s, d, n);
  };
  cvt(wq, wq_h, CC * CC);
  cvt(wkv, wkv_h, 2 * CC * CC);
  cvt(w_proj, wpr_h, CC * CC);
  cvt(w_fc1, wf1_h, 4 * CC * CC);
  cvt(w_fc2, wf2_h, 4 * CC * CC);

  // ---- LN1 + roll + window partition ----
  ln_part_kernel<<<SROWS / 8, 256, 0, stream>>>(x, y, q_mask, s_mask, pro_pred,
                                                norm1_g, norm1_b, s_buf, qb_s,
                                                sb_s, ppc_s, hor_s);

  // ---- Q and KV projections (f16 WMMA, f32 accumulate) ----
  gemm_wmma<true, false, false><<<dim3(SROWS / 64, CC / 128), 256, 0, stream>>>(
      s_buf, wq_h, bq, nullptr, q_buf, SROWS, CC, CC);
  gemm_wmma<true, false, false><<<dim3(SROWS / 64, 2 * CC / 128), 256, 0, stream>>>(
      s_buf, wkv_h, bkv, nullptr, kv_buf, SROWS, 2 * CC, CC);

  // ---- attention per (window, head) ----
  attn_kernel<<<dim3(NWIN, NHEAD), 256, 0, stream>>>(
      q_buf, kv_buf, qb_s, sb_s, ppc_s, hor_s, w_pq, b_pq, w_ps, b_ps, attn_o);

  // ---- output projection ----
  gemm_wmma<false, false, false><<<dim3(MTOT / 64, CC / 128), 256, 0, stream>>>(
      attn_o, wpr_h, b_proj, nullptr, proj_o, MTOT, CC, CC);

  // ---- window-reverse + residual + LN2 ----
  fuse_resid_ln2<<<MTOT / 8, 256, 0, stream>>>(x, proj_o, norm2_g, norm2_b, x1, xm);

  // ---- MLP: fc1 (+GELU) then fc2 (+final residual) -> d_out (f32) ----
  gemm_wmma<true, true, false><<<dim3(MTOT / 64, 4 * CC / 128), 256, 0, stream>>>(
      xm, wf1_h, b_fc1, nullptr, h_buf, MTOT, 4 * CC, CC);
  gemm_wmma<false, false, true><<<dim3(MTOT / 64, CC / 128), 256, 0, stream>>>(
      h_buf, wf2_h, b_fc2, x1, (float*)d_out, MTOT, CC, 4 * CC);
}
